// ConvNetLayer_22239340658714
// MI455X (gfx1250) — matrix operand order repaired and checked by reading, not compiled, not run on hardware
//
#include <hip/hip_runtime.h>
#include <hip/hip_bf16.h>

#define HID 256
#define LDE 264  // padded LDS row stride (bf16 elements): 528B rows -> 16B-aligned frags, spread banks

typedef __attribute__((ext_vector_type(16))) __bf16        v16bf;
typedef __attribute__((ext_vector_type(8)))  float         v8f;
typedef __attribute__((ext_vector_type(4)))  unsigned int  u32x4;

union Frag16 {
    u32x4 u[2];
    v16bf v;
};

__device__ __forceinline__ unsigned short f2bf(float f) {
    unsigned int u = __float_as_uint(f);
    u += 0x7FFFu + ((u >> 16) & 1u);   // round-to-nearest-even
    return (unsigned short)(u >> 16);
}

// ---------------------------------------------------------------------------
// Weight prep: f32 W[o][k] -> bf16 B-fragment-major blob.
// Per (matrix m, kstep ks, ntile nt): 32x16 bf16 fragment = 32 lanes x 16 elems.
// lane = (o&15) | ((k>>3)&1)<<4; elem = (k&7) | ((k>>4)&1)<<3  (ISA 7.12.2 16-bit B layout, wave32)
// ---------------------------------------------------------------------------
__global__ __launch_bounds__(256) void prep_weights(
    const float* __restrict__ Uw, const float* __restrict__ Bw,
    const float* __restrict__ Cw, const float* __restrict__ Vw,
    const float* __restrict__ Aw, unsigned short* __restrict__ wb) {
    int t = blockIdx.x * blockDim.x + threadIdx.x;
    if (t >= 5 * HID * HID) return;
    int m = t >> 16;
    int r = t & 0xFFFF;
    int o = r >> 8;
    int k = r & 255;
    const float* W = (m == 0) ? Uw : (m == 1) ? Bw : (m == 2) ? Cw : (m == 3) ? Vw : Aw;
    float val = W[o * HID + k];
    int ks = k >> 5;
    int kk = k & 31;
    int nt = o >> 4;
    int lane = (o & 15) | (((kk >> 3) & 1) << 4);
    int elem = (kk & 7) | ((kk >> 4) << 3);
    size_t off = ((size_t)((m * 8 + ks) * 16 + nt)) * 512 + (size_t)lane * 16 + elem;
    wb[off] = f2bf(val);
}

// Stage a 64 x 256 f32 tile into LDS as bf16 with padded stride.
__device__ __forceinline__ void stage_tile_bf16(const float* __restrict__ src,
                                                int rowbase, int nrows,
                                                unsigned short* As) {
    for (int i = threadIdx.x; i < 64 * 64; i += blockDim.x) {
        int r  = i >> 6;
        int c4 = (i & 63) << 2;
        int g  = rowbase + r;
        float4 v = make_float4(0.f, 0.f, 0.f, 0.f);
        if (g < nrows) v = *(const float4*)(src + (size_t)g * HID + c4);
        unsigned int p0 = (unsigned int)f2bf(v.x) | ((unsigned int)f2bf(v.y) << 16);
        unsigned int p1 = (unsigned int)f2bf(v.z) | ((unsigned int)f2bf(v.w) << 16);
        unsigned int* d = (unsigned int*)(As + r * LDE + c4);
        d[0] = p0;
        d[1] = p1;
    }
}

// A-fragment (16x32 bf16, ISA 7.12.2): lane l<16: row=l, K = k0+{0..7,16..23};
// lane l>=16: row=l-16, K = k0+{8..15,24..31}.
__device__ __forceinline__ v16bf load_afrag(const unsigned short* As, int mt, int ks, int lane) {
    int row  = mt * 16 + (lane & 15);
    int sel  = (lane >> 4) & 1;
    int eoff = row * LDE + ks * 32 + sel * 8;
    Frag16 f;
    f.u[0] = *(const u32x4*)(As + eoff);        // ds_load_b128
    f.u[1] = *(const u32x4*)(As + eoff + 16);   // ds_load_b128
    return f.v;
}

// B-fragment from pre-swizzled blob: one 32B coalesced read per lane.
__device__ __forceinline__ v16bf load_bfrag(const unsigned short* __restrict__ wb,
                                            int m, int ks, int nt, int lane) {
    size_t off = ((size_t)((m * 8 + ks) * 16 + nt)) * 512 + (size_t)lane * 16;
    Frag16 f;
    f.u[0] = *(const u32x4*)(wb + off);
    f.u[1] = *(const u32x4*)(wb + off + 8);
    return f.v;
}

// ---------------------------------------------------------------------------
// Node-side: 4 waves, wave w computes atom @ {U,B,C,V}[w]^T + bias over a
// 64-row tile; single LDS staging reused by all 4 weight matrices.
// ---------------------------------------------------------------------------
__global__ __launch_bounds__(128) void atom_gemm(
    const float* __restrict__ x, const unsigned short* __restrict__ wb,
    const float* __restrict__ bU, const float* __restrict__ bB,
    const float* __restrict__ bC, const float* __restrict__ bV,
    float* __restrict__ Ux, float* __restrict__ Bx,
    float* __restrict__ Cx, float* __restrict__ Vx, int N) {
    __shared__ unsigned short As[64 * LDE];
    int rowbase = blockIdx.x * 64;
    stage_tile_bf16(x, rowbase, N, As);
    __syncthreads();

    int w    = threadIdx.x >> 5;
    int lane = threadIdx.x & 31;
    const float* bias = (w == 0) ? bU : (w == 1) ? bB : (w == 2) ? bC : bV;
    float*       out  = (w == 0) ? Ux : (w == 1) ? Bx : (w == 2) ? Cx : Vx;

    for (int nt = 0; nt < 16; ++nt) {
        v8f acc[4];
#pragma unroll
        for (int mt = 0; mt < 4; ++mt) acc[mt] = (v8f){0, 0, 0, 0, 0, 0, 0, 0};
#pragma unroll
        for (int ks = 0; ks < 8; ++ks) {
            v16bf bf = load_bfrag(wb, w, ks, nt, lane);
#pragma unroll
            for (int mt = 0; mt < 4; ++mt) {
                v16bf af = load_afrag(As, mt, ks, lane);
                acc[mt] = __builtin_amdgcn_wmma_f32_16x16x32_bf16(
                    false, af, false, bf, (short)0, acc[mt], false, false);
            }
        }
        int col   = nt * 16 + (lane & 15);
        float b   = bias[col];
        int mrow  = (lane >> 4) << 3;
#pragma unroll
        for (int mt = 0; mt < 4; ++mt) {
#pragma unroll
            for (int r = 0; r < 8; ++r) {
                int g = rowbase + mt * 16 + r + mrow;
                if (g < N) out[(size_t)g * HID + col] = acc[mt][r] + b;
            }
        }
    }
}

// ---------------------------------------------------------------------------
// Edge-side: bond @ Aw^T with fused gather/activation/scatter epilogue.
// 4 waves; wave w covers column tiles [4w, 4w+4).
// ---------------------------------------------------------------------------
__global__ __launch_bounds__(128) void bond_gemm(
    const float* __restrict__ bond, const unsigned short* __restrict__ wb,
    const float* __restrict__ Ab, const int* __restrict__ esrc,
    const int* __restrict__ edst, const float* __restrict__ Bx,
    const float* __restrict__ Cx, const float* __restrict__ Vx,
    float* __restrict__ Agg, float* __restrict__ bond_out, int E) {
    __shared__ unsigned short As[64 * LDE];
    __shared__ int s_src[64];
    __shared__ int s_dst[64];
    int rowbase = blockIdx.x * 64;
    stage_tile_bf16(bond, rowbase, E, As);
    if (threadIdx.x < 64) {
        int g = rowbase + threadIdx.x;
        s_src[threadIdx.x] = (g < E) ? esrc[g] : 0;
        s_dst[threadIdx.x] = (g < E) ? edst[g] : 0;
    }
    __syncthreads();

    int w    = threadIdx.x >> 5;
    int lane = threadIdx.x & 31;

    for (int nti = 0; nti < 4; ++nti) {
        int nt = w * 4 + nti;
        v8f acc[4];
#pragma unroll
        for (int mt = 0; mt < 4; ++mt) acc[mt] = (v8f){0, 0, 0, 0, 0, 0, 0, 0};
#pragma unroll
        for (int ks = 0; ks < 8; ++ks) {
            v16bf bf = load_bfrag(wb, 4, ks, nt, lane);
#pragma unroll
            for (int mt = 0; mt < 4; ++mt) {
                v16bf af = load_afrag(As, mt, ks, lane);
                acc[mt] = __builtin_amdgcn_wmma_f32_16x16x32_bf16(
                    false, af, false, bf, (short)0, acc[mt], false, false);
            }
        }
        int col  = nt * 16 + (lane & 15);
        float ab = Ab[col];
        int mrow = (lane >> 4) << 3;
#pragma unroll
        for (int mt = 0; mt < 4; ++mt) {
#pragma unroll
            for (int r = 0; r < 8; ++r) {
                int lrow = mt * 16 + r + mrow;
                int grow = rowbase + lrow;
                if (grow < E) {
                    int s = s_src[lrow];
                    int d = s_dst[lrow];
                    float g = acc[mt][r] + ab + Bx[(size_t)s * HID + col] +
                              Cx[(size_t)d * HID + col];
                    bond_out[(size_t)grow * HID + col] = fmaxf(g, 0.f);
                    float sg  = 1.f / (1.f + __expf(-g));
                    float msg = sg * Vx[(size_t)d * HID + col];
                    atomicAdd(&Agg[(size_t)s * HID + col], msg);
                }
            }
        }
    }
}

__global__ __launch_bounds__(256) void relu_out(const float* __restrict__ Agg,
                                                float* __restrict__ out, size_t n) {
    size_t i = (size_t)blockIdx.x * blockDim.x + threadIdx.x;
    if (i < n) out[i] = fmaxf(Agg[i], 0.f);
}

extern "C" void kernel_launch(void* const* d_in, const int* in_sizes, int n_in,
                              void* d_out, int out_size, void* d_ws, size_t ws_size,
                              hipStream_t stream) {
    const float* atom = (const float*)d_in[0];
    const float* bond = (const float*)d_in[1];
    const int*   esrc = (const int*)d_in[2];
    const int*   edst = (const int*)d_in[3];
    const float* Uw = (const float*)d_in[4];
    const float* Ub = (const float*)d_in[5];
    const float* Vw = (const float*)d_in[6];
    const float* Vb = (const float*)d_in[7];
    const float* Aw = (const float*)d_in[8];
    const float* Ab = (const float*)d_in[9];
    const float* Bw = (const float*)d_in[10];
    const float* Bb = (const float*)d_in[11];
    const float* Cw = (const float*)d_in[12];
    const float* Cb = (const float*)d_in[13];

    int N = in_sizes[0] / HID;
    int E = in_sizes[2];

    unsigned short* wb = (unsigned short*)d_ws;                         // 640 KB bf16 weight blobs
    float* Ux = (float*)((char*)d_ws + (size_t)5 * HID * HID * 2);      // pre-activation + agg
    float* Bx = Ux + (size_t)N * HID;
    float* Cx = Bx + (size_t)N * HID;
    float* Vx = Cx + (size_t)N * HID;

    float* out_atom = (float*)d_out;
    float* out_bond = out_atom + (size_t)N * HID;

    prep_weights<<<(5 * HID * HID + 255) / 256, 256, 0, stream>>>(Uw, Bw, Cw, Vw, Aw, wb);
    atom_gemm<<<(N + 63) / 64, 128, 0, stream>>>(atom, wb, Ub, Bb, Cb, Vb, Ux, Bx, Cx, Vx, N);
    bond_gemm<<<(E + 63) / 64, 128, 0, stream>>>(bond, wb, Ab, esrc, edst, Bx, Cx, Vx, Ux,
                                                 out_bond, E);
    size_t n = (size_t)N * HID;
    relu_out<<<(unsigned)((n + 255) / 256), 256, 0, stream>>>(Ux, out_atom, n);
}
